// DetContrastiveLoss_72636486910305
// MI455X (gfx1250) — matrix operand order repaired and checked by reading, not compiled, not run on hardware
//
#include <hip/hip_runtime.h>
#include <hip/hip_bf16.h>
#include <hip/hip_fp16.h>

typedef __attribute__((ext_vector_type(16))) _Float16 v16h;
typedef __attribute__((ext_vector_type(8)))  float    v8f;

#define M_TOTAL 1024          // B*N
#define CCH     256           // channels / K
#define HH      360
#define WW      360
#define NEG_FILL_F (-1.0e9f)

// order-preserving float -> u32 key (so integer ds_max works for floats)
__device__ __forceinline__ unsigned fkey(float f) {
    unsigned u = __float_as_uint(f);
    return (u & 0x80000000u) ? ~u : (u | 0x80000000u);
}
__device__ __forceinline__ float funkey(unsigned k) {
    return __uint_as_float((k & 0x80000000u) ? (k ^ 0x80000000u) : ~k);
}

// ---------------------------------------------------------------------------
// Kernel 1: gather feature vector per box, L2-normalize, emit f16 row + category
// grid = 1024 blocks (one per box), block = 256 threads (one per channel)
// ---------------------------------------------------------------------------
__global__ void k_gather_normalize(const float* __restrict__ spatial,
                                   const float* __restrict__ gt_boxes,
                                   _Float16* __restrict__ fn,
                                   int* __restrict__ cat) {
    __shared__ float red[256];
    __shared__ int s_cx, s_cy, s_cat;

    const int row = blockIdx.x;          // b*256 + n
    const int b   = row >> 8;
    const int tid = threadIdx.x;

    if (tid == 0) {
        const float* gb = gt_boxes + row * 9;
        float fx = (gb[0] + 59.9f) * (360.0f / 119.8f);
        float fy = (gb[1] + 59.9f) * (360.0f / 119.8f);
        fx = fminf(fmaxf(fx, 0.0f), 359.0f);
        fy = fminf(fmaxf(fy, 0.0f), 359.0f);
        s_cx = (int)fx;
        s_cy = (int)fy;
        int dyn = (gb[7] != 0.0f) ? 1 : 0;
        int cls = (int)gb[8];
        s_cat = dyn * 3 + cls;
    }
    __syncthreads();

    const int cx = s_cx, cy = s_cy;
    const float v = spatial[(((size_t)b * CCH + tid) * HH + cy) * WW + cx];

    red[tid] = v * v;
    __syncthreads();
    for (int s = 128; s > 0; s >>= 1) {
        if (tid < s) red[tid] += red[tid + s];
        __syncthreads();
    }
    const float norm = fmaxf(sqrtf(red[0]), 1e-12f);
    fn[(size_t)row * CCH + tid] = (_Float16)(v / norm);

    if (tid == 0) cat[row] = s_cat;
}

// ---------------------------------------------------------------------------
// Kernel 2: sim = (fn fn^T)/T via v_wmma_f32_16x16x32_f16, fused per-row
// category maxima (6 cats) into LDS via ds integer atomic-max.
// grid = 64 blocks (16-row strips), block = 256 threads = 8 waves.
// Each wave: preload 8 A-fragments (16 rows x K=256), loop over 8 col tiles.
// ---------------------------------------------------------------------------
__global__ void k_sim_rowmax(const _Float16* __restrict__ fn,
                             const int* __restrict__ colcat,
                             float* __restrict__ rowmax) {
    __shared__ unsigned rmx[16 * 6];

    const int tid    = threadIdx.x;
    const int lane   = tid & 31;
    const int wave   = tid >> 5;          // 0..7
    const int lane15 = lane & 15;
    const int half   = lane >> 4;         // 0 or 1
    const int m0     = blockIdx.x * 16;

    if (tid < 96) rmx[tid] = fkey(NEG_FILL_F);
    __syncthreads();

    const uint4* fnq = reinterpret_cast<const uint4*>(fn); // 8 halves per uint4

    union Frag { v16h h; uint4 q[2]; };

    // --- preload A fragments: rows m0..m0+15, all K (8 chunks of 32) ---
    // A layout (16-bit 16x32): half 0 lanes hold K={0..7,16..23},
    //                          half 1 lanes hold K={8..15,24..31}
    Frag a[8];
    {
        const int m = m0 + lane15;
        const int abase = m * (CCH / 8) + half; // (m*256 + half*8)/8
        #pragma unroll
        for (int kc = 0; kc < 8; ++kc) {
            a[kc].q[0] = fnq[abase + kc * 4];       // K = k0 + half*8 + 0..7
            a[kc].q[1] = fnq[abase + kc * 4 + 2];   // K = k0 + half*8 + 16..23
        }
    }

    // --- sweep column tiles; each wave does tiles wave, wave+8, ... ---
    for (int t = 0; t < 8; ++t) {
        const int n0 = (wave + t * 8) * 16;
        const int n  = n0 + lane15;
        // B = fn^T : lane n reads fn row n; half selects K {0..7,16..23} / {8..15,24..31}
        const int bbase = n * (CCH / 8) + half; // (n*256 + half*8)/8

        v8f acc = {};
        #pragma unroll
        for (int kc = 0; kc < 8; ++kc) {
            Frag bfr;
            bfr.q[0] = fnq[bbase + kc * 4];
            bfr.q[1] = fnq[bbase + kc * 4 + 2];
            acc = __builtin_amdgcn_wmma_f32_16x16x32_f16(
                false, a[kc].h, false, bfr.h, (short)0, acc, false, false);
        }

        const int g = colcat[n];
        #pragma unroll
        for (int r = 0; r < 8; ++r) {
            const int rl = (lane < 16) ? r : (r + 8);
            const float v = acc[r] * 10.0f;   // / TEMPERATURE
            atomicMax(&rmx[rl * 6 + g], fkey(v));
        }
    }
    __syncthreads();

    if (tid < 96) {
        const int rl = tid / 6, g = tid % 6;
        rowmax[(size_t)(m0 + rl) * 6 + g] = funkey(rmx[tid]);
    }
}

// ---------------------------------------------------------------------------
// Kernel 3: per-group hinge means -> scalar loss. 1 block x 256 threads.
// ---------------------------------------------------------------------------
__global__ void k_finish(const float* __restrict__ rowmax,
                         const int* __restrict__ cat,
                         float* __restrict__ out) {
    __shared__ float gsum[6];
    __shared__ int   gcnt[6];
    const int tid = threadIdx.x;
    if (tid < 6) { gsum[tid] = 0.0f; gcnt[tid] = 0; }
    __syncthreads();

    for (int i = tid; i < M_TOTAL; i += 256) {
        const int cg  = cat[i];          // anchor group of this row
        const int d   = cg / 3;
        const int c   = cg - d * 3;
        const int ob  = (1 - d) * 3;     // opposite-dyn base
        const float mp = rowmax[(size_t)i * 6 + ob + c];
        const float mn = fmaxf(rowmax[(size_t)i * 6 + ob + ((c + 1) % 3)],
                               rowmax[(size_t)i * 6 + ob + ((c + 2) % 3)]);
        const float hinge = fmaxf(0.0f, 0.2f + mn - mp);
        atomicAdd(&gsum[cg], hinge);
        atomicAdd(&gcnt[cg], 1);
    }
    __syncthreads();

    if (tid == 0) {
        float total = 0.0f, count = 0.0f;
        for (int ad = 0; ad < 2; ++ad) {
            for (int c = 0; c < 3; ++c) {
                const int ob = (1 - ad) * 3;
                const int na = gcnt[ad * 3 + c];
                const int np = gcnt[ob + c];
                const int nn = gcnt[ob + (c + 1) % 3] + gcnt[ob + (c + 2) % 3];
                if (na > 0 && np > 0 && nn > 0) {
                    total += gsum[ad * 3 + c] / (float)na;
                    count += 1.0f;
                }
            }
        }
        out[0] = (count > 0.0f) ? (total / count) : 0.0f;
    }
}

// ---------------------------------------------------------------------------
extern "C" void kernel_launch(void* const* d_in, const int* in_sizes, int n_in,
                              void* d_out, int out_size, void* d_ws, size_t ws_size,
                              hipStream_t stream) {
    const float* spatial  = (const float*)d_in[0];
    const float* gt_boxes = (const float*)d_in[1];
    float* out = (float*)d_out;

    char* ws = (char*)d_ws;
    _Float16* fn     = (_Float16*)ws;                          // 1024*256*2 = 524288 B
    int*      cat    = (int*)(ws + 524288);                    // 4096 B
    float*    rowmax = (float*)(ws + 524288 + 4096);           // 1024*6*4 = 24576 B

    k_gather_normalize<<<M_TOTAL, 256, 0, stream>>>(spatial, gt_boxes, fn, cat);
    k_sim_rowmax<<<M_TOTAL / 16, 256, 0, stream>>>(fn, cat, rowmax);
    k_finish<<<1, 256, 0, stream>>>(rowmax, cat, out);
}
